// GAT_49014166782118
// MI455X (gfx1250) — compile-verified, matrix operand order checked by reading, not run on hardware
//
#include <hip/hip_runtime.h>
#include <hip/hip_bf16.h>

// ---------------------------------------------------------------------------
// GAT 2-layer forward for MI455X (gfx1250, wave32, WMMA).
// GEMMs via v_wmma_f32_16x16x32_bf16 with fragment-contiguous LDS layouts
// (2x ds_load_b128 per operand fragment, fragments preloaded so WMMAs issue
// back-to-back); edge softmax/aggregation via L2 atomics.
// ---------------------------------------------------------------------------

typedef __bf16 bf16t;
typedef __attribute__((ext_vector_type(16))) __bf16 v16bf;
typedef __attribute__((ext_vector_type(8)))  __bf16 v8bf;
typedef __attribute__((ext_vector_type(4)))  __bf16 v4bf;
typedef __attribute__((ext_vector_type(8)))  float  v8f;

#define NEG_SLOPE 0.2f
#define EPSF 1e-16f

__device__ __forceinline__ v8bf ld_v8bf(const bf16t* p) {
  return *(const v8bf*)__builtin_assume_aligned(p, 16);
}
__device__ __forceinline__ void st_v4bf(bf16t* p, v4bf v) {
  *(v4bf*)__builtin_assume_aligned(p, 8) = v;
}

// monotone float<->uint mapping so unsigned atomicMax == float max
__device__ __forceinline__ unsigned fflip(float f) {
  unsigned u = __float_as_uint(f);
  return (u & 0x80000000u) ? ~u : (u | 0x80000000u);
}
__device__ __forceinline__ float funflip(unsigned u) {
  return __uint_as_float((u & 0x80000000u) ? (u ^ 0x80000000u) : ~u);
}

// ---------------------------------------------------------------------------
// WMMA GEMM: H[M, outc] = X[M, K] * W[K, outc]   (f32 in, bf16 WMMA, f32 acc)
// Block = 256 threads = 8 waves; each wave owns 16 rows x (NTILES*16) cols.
// K chunked by 64 through LDS (2 wmma K-steps of 32 per chunk).
// ---------------------------------------------------------------------------
template<int K, int NTILES>
__global__ __launch_bounds__(256)
void gemm_wmma_kernel(const float* __restrict__ X, const float* __restrict__ W,
                      float* __restrict__ H, int M, int outc)
{
  constexpr int NCOLS = NTILES * 16;
  constexpr int LDA = 72;   // bf16 elems/row; 144 B (16-aligned, odd 16-banks)
  constexpr int LDB = 72;
  __shared__ __align__(16) bf16t sA[128 * LDA];     // [row][k]
  __shared__ __align__(16) bf16t sBt[NCOLS * LDB];  // [n][k]  (W transposed)

  const int tid  = threadIdx.x;
  const int lane = tid & 31;
  const int wid  = tid >> 5;
  const int r0   = blockIdx.x * 128;

  v8f acc[NTILES] = {};

  const int arow = tid >> 1;         // 0..127
  const int acol = (tid & 1) * 32;   // 0 or 32
  const int grow = r0 + arow;

  const int mrow = lane & 15;
  const int kb   = (lane >> 4) * 8;  // ISA 16-bit operand layout: K-halves 0/8

  for (int kc = 0; kc < K; kc += 64) {
    // ---- stage A chunk (f32 -> bf16, packed 64-bit LDS stores) ----
    if (grow < M) {
      const float4* xp = (const float4*)(X + (size_t)grow * K + kc + acol);
#pragma unroll
      for (int q = 0; q < 8; ++q) {
        float4 v = xp[q];
        st_v4bf(&sA[arow * LDA + acol + q * 4],
                (v4bf){(bf16t)v.x, (bf16t)v.y, (bf16t)v.z, (bf16t)v.w});
      }
    } else {
#pragma unroll
      for (int q = 0; q < 8; ++q)
        st_v4bf(&sA[arow * LDA + acol + q * 4], (v4bf){(bf16t)0.f, (bf16t)0.f, (bf16t)0.f, (bf16t)0.f});
    }
    // ---- stage B chunk transposed (zero-pad cols beyond outc) ----
    for (int i = tid; i < NCOLS * 16; i += 256) {
      int n  = i >> 4;
      int k4 = (i & 15) * 4;
      v4bf w;
#pragma unroll
      for (int j = 0; j < 4; ++j)
        w[j] = (bf16t)((n < outc) ? W[(size_t)(kc + k4 + j) * outc + n] : 0.0f);
      st_v4bf(&sBt[n * LDB + k4], w);
    }
    __syncthreads();

    if (kc + 64 < K && grow < M)
      __builtin_prefetch(X + (size_t)grow * K + kc + 64 + acol, 0, 0);

#pragma unroll
    for (int ks = 0; ks < 2; ++ks) {
      const int kk0 = ks * 32;
      // Preload A fragment + ALL B fragments, then issue WMMAs back-to-back
      // (single dscnt wait, matrix pipe stays busy).
      const bf16t* ap = &sA[(wid * 16 + mrow) * LDA + kk0 + kb];
      v16bf afrag = __builtin_shufflevector(
          ld_v8bf(ap), ld_v8bf(ap + 16),
          0, 1, 2, 3, 4, 5, 6, 7, 8, 9, 10, 11, 12, 13, 14, 15);
      v16bf bfr[NTILES];
#pragma unroll
      for (int t = 0; t < NTILES; ++t) {
        const bf16t* bp = &sBt[(t * 16 + mrow) * LDB + kk0 + kb];
        bfr[t] = __builtin_shufflevector(
            ld_v8bf(bp), ld_v8bf(bp + 16),
            0, 1, 2, 3, 4, 5, 6, 7, 8, 9, 10, 11, 12, 13, 14, 15);
      }
#pragma unroll
      for (int t = 0; t < NTILES; ++t)
        acc[t] = __builtin_amdgcn_wmma_f32_16x16x32_bf16(
            false, afrag, false, bfr[t], (short)0, acc[t], false, false);
    }
    __syncthreads();
  }

  // C/D layout: lanes 0-15 -> M=r, lanes 16-31 -> M=8+r; N = lane&15
  const int ncol = lane & 15;
  const int moff = (lane >> 4) * 8;
  const int m0   = r0 + wid * 16;
#pragma unroll
  for (int t = 0; t < NTILES; ++t) {
#pragma unroll
    for (int r = 0; r < 8; ++r) {
      int row = m0 + moff + r;
      int col = t * 16 + ncol;
      if (row < M && col < outc)
        H[(size_t)row * outc + col] = acc[t][r];
    }
  }
}

// a_src[n,h] = <h[n,h,:], att_src[h,:]>, same for dst. One thread per (n,head).
__global__ void attn_prep_kernel(const float* __restrict__ H,
                                 const float* __restrict__ att_src,
                                 const float* __restrict__ att_dst,
                                 float* __restrict__ asrc, float* __restrict__ adst,
                                 int N, int heads, int C)
{
  int idx = blockIdx.x * blockDim.x + threadIdx.x;
  if (idx >= N * heads) return;
  int n = idx / heads, h = idx % heads;
  const float* hp = H + (size_t)n * heads * C + (size_t)h * C;
  const float* as = att_src + h * C;
  const float* ad = att_dst + h * C;
  float s = 0.f, d = 0.f;
  for (int c = 0; c < C; ++c) { float v = hp[c]; s += v * as[c]; d += v * ad[c]; }
  asrc[idx] = s;
  adst[idx] = d;
}

__global__ void init_kernel(unsigned* __restrict__ m, float* __restrict__ den,
                            float* __restrict__ acc, int nm, int nacc)
{
  int i = blockIdx.x * blockDim.x + threadIdx.x;
  if (i < nm) { m[i] = 0x007FFFFFu; /* fflip(-inf) */ den[i] = 0.f; }
  if (i < nacc) acc[i] = 0.f;
}

template<int HE>
__global__ void edge_max_kernel(const long long* __restrict__ ei, int E, int N,
                                const float* __restrict__ asrc,
                                const float* __restrict__ adst,
                                unsigned* __restrict__ m)
{
  int e = blockIdx.x * blockDim.x + threadIdx.x;
  if (e >= E + N) return;
  int s, d;
  if (e < E) { s = (int)ei[e]; d = (int)ei[(size_t)E + e]; }
  else       { s = d = e - E; }               // self loop
#pragma unroll
  for (int h = 0; h < HE; ++h) {
    float v = asrc[s * HE + h] + adst[d * HE + h];
    v = (v > 0.f) ? v : NEG_SLOPE * v;
    atomicMax(&m[d * HE + h], fflip(v));
  }
}

// ex = exp(e - max); accumulate denom and unnormalized messages (ex * h[src]).
template<int HE, int C>
__global__ void edge_scatter_kernel(const long long* __restrict__ ei, int E, int N,
                                    const float* __restrict__ asrc,
                                    const float* __restrict__ adst,
                                    const unsigned* __restrict__ m,
                                    const float* __restrict__ H,
                                    float* __restrict__ den,
                                    float* __restrict__ acc)
{
  int e = blockIdx.x * blockDim.x + threadIdx.x;
  if (e >= E + N) return;
  int s, d;
  if (e < E) { s = (int)ei[e]; d = (int)ei[(size_t)E + e]; }
  else       { s = d = e - E; }
#pragma unroll
  for (int h = 0; h < HE; ++h) {
    float v = asrc[s * HE + h] + adst[d * HE + h];
    v = (v > 0.f) ? v : NEG_SLOPE * v;
    float ex = __expf(v - funflip(m[d * HE + h]));
    atomicAdd(&den[d * HE + h], ex);
    const float* hs = H + ((size_t)s * HE + h) * C;
    float* ap = acc + ((size_t)d * HE + h) * C;
#pragma unroll
    for (int c = 0; c < C; ++c) atomicAdd(&ap[c], ex * hs[c]);
  }
}

// layer-1 epilogue: normalize, +bias, ReLU  (heads=8, C=8 -> 64 features)
__global__ void norm_relu_kernel(const float* __restrict__ acc,
                                 const float* __restrict__ den,
                                 const float* __restrict__ bias,
                                 float* __restrict__ out, int N)
{
  int i = blockIdx.x * blockDim.x + threadIdx.x;
  if (i >= N * 64) return;
  int k = i & 63;
  int n = i >> 6;
  int h = k >> 3;
  float v = acc[i] / (den[n * 8 + h] + EPSF) + bias[k];
  out[i] = fmaxf(v, 0.f);
}

// layer-2 epilogue: normalize, +bias, log_softmax over 40 classes
__global__ void finalize_kernel(const float* __restrict__ acc,
                                const float* __restrict__ den,
                                const float* __restrict__ bias,
                                float* __restrict__ out, int N)
{
  int n = blockIdx.x * blockDim.x + threadIdx.x;
  if (n >= N) return;
  float dinv = 1.f / (den[n] + EPSF);
  float v[40];
  float mx = -3.4e38f;
#pragma unroll
  for (int c = 0; c < 40; ++c) {
    v[c] = acc[(size_t)n * 40 + c] * dinv + bias[c];
    mx = fmaxf(mx, v[c]);
  }
  float sum = 0.f;
#pragma unroll
  for (int c = 0; c < 40; ++c) sum += __expf(v[c] - mx);
  float lse = mx + __logf(sum);
#pragma unroll
  for (int c = 0; c < 40; ++c) out[(size_t)n * 40 + c] = v[c] - lse;
}

// ---------------------------------------------------------------------------
extern "C" void kernel_launch(void* const* d_in, const int* in_sizes, int n_in,
                              void* d_out, int out_size, void* d_ws, size_t ws_size,
                              hipStream_t stream)
{
  const float*     x   = (const float*)d_in[0];
  const long long* ei  = (const long long*)d_in[1];   // int64 edge_index [2,E]
  const float*     W1  = (const float*)d_in[2];
  const float*     as1 = (const float*)d_in[3];
  const float*     ad1 = (const float*)d_in[4];
  const float*     b1  = (const float*)d_in[5];
  const float*     W2  = (const float*)d_in[6];
  const float*     as2 = (const float*)d_in[7];
  const float*     ad2 = (const float*)d_in[8];
  const float*     b2  = (const float*)d_in[9];

  const int FIN = 512;
  const int N  = in_sizes[0] / FIN;
  const int E  = in_sizes[1] / 2;
  const int ET = E + N;

  // workspace layout (peak N*160 floats = 64 MB @ N=100k; fits in L2)
  float*    h1    = (float*)d_ws;                       // N*64
  float*    acc1  = h1    + (size_t)N * 64;             // N*64 (reused as h2pre)
  float*    asrc1 = acc1  + (size_t)N * 64;             // N*8
  float*    adst1 = asrc1 + (size_t)N * 8;              // N*8
  unsigned* m1    = (unsigned*)(adst1 + (size_t)N * 8); // N*8
  float*    den1  = (float*)m1 + (size_t)N * 8;         // N*8

  const int TB = 256;

  // ---------------- Layer 1: heads=8, C=8, concat ----------------
  gemm_wmma_kernel<512, 4><<<dim3((N + 127) / 128), dim3(TB), 0, stream>>>(x, W1, h1, N, 64);
  attn_prep_kernel<<<dim3((N * 8 + TB - 1) / TB), dim3(TB), 0, stream>>>(h1, as1, ad1, asrc1, adst1, N, 8, 8);
  init_kernel<<<dim3((N * 64 + TB - 1) / TB), dim3(TB), 0, stream>>>(m1, den1, acc1, N * 8, N * 64);
  edge_max_kernel<8><<<dim3((ET + TB - 1) / TB), dim3(TB), 0, stream>>>(ei, E, N, asrc1, adst1, m1);
  edge_scatter_kernel<8, 8><<<dim3((ET + TB - 1) / TB), dim3(TB), 0, stream>>>(ei, E, N, asrc1, adst1, m1, h1, den1, acc1);
  norm_relu_kernel<<<dim3((N * 64 + TB - 1) / TB), dim3(TB), 0, stream>>>(acc1, den1, b1, h1, N);

  // ---------------- Layer 2: heads=1, C=40 ----------------
  float*    h2    = acc1;   // acc1 free after norm_relu
  float*    asrc2 = asrc1;
  float*    adst2 = adst1;
  unsigned* m2    = m1;
  float*    den2  = den1;
  float*    acc2  = h1;     // h1 free after gemm2+attn_prep consume it

  gemm_wmma_kernel<64, 3><<<dim3((N + 127) / 128), dim3(TB), 0, stream>>>(h1, W2, h2, N, 40);
  attn_prep_kernel<<<dim3((N + TB - 1) / TB), dim3(TB), 0, stream>>>(h2, as2, ad2, asrc2, adst2, N, 1, 40);
  init_kernel<<<dim3((N * 40 + TB - 1) / TB), dim3(TB), 0, stream>>>(m2, den2, acc2, N, N * 40);
  edge_max_kernel<1><<<dim3((ET + TB - 1) / TB), dim3(TB), 0, stream>>>(ei, E, N, asrc2, adst2, m2);
  edge_scatter_kernel<1, 40><<<dim3((ET + TB - 1) / TB), dim3(TB), 0, stream>>>(ei, E, N, asrc2, adst2, m2, h2, den2, acc2);
  finalize_kernel<<<dim3((N + TB - 1) / TB), dim3(TB), 0, stream>>>(acc2, den2, b2, (float*)d_out, N);
}
